// CGCNN_47974784696406
// MI455X (gfx1250) — matrix operand order repaired and checked by reading, not compiled, not run on hardware
//
#include <hip/hip_runtime.h>

// ---------------------------------------------------------------------------
// CGCNN inference for MI455X (gfx1250, wave32, WMMA bf16)
// ---------------------------------------------------------------------------

typedef __attribute__((ext_vector_type(16))) __bf16 v16bf;
typedef __attribute__((ext_vector_type(8)))  float  v8f;

union Frag {
    unsigned u[8];
    v16bf    v;
};

__device__ __forceinline__ unsigned short f32_to_bf16(float f) {
    unsigned u = __float_as_uint(f);
    u += 0x7FFFu + ((u >> 16) & 1u);   // round-to-nearest-even
    return (unsigned short)(u >> 16);
}

__device__ __forceinline__ unsigned pk_bf16(float lo, float hi) {
    return (unsigned)f32_to_bf16(lo) | ((unsigned)f32_to_bf16(hi) << 16);
}

// ---------------------------------------------------------------------------
// generic zero-fill
// ---------------------------------------------------------------------------
__global__ void k_zero(float* __restrict__ p, int n) {
    int i = blockIdx.x * blockDim.x + threadIdx.x;
    if (i < n) p[i] = 0.0f;
}

// ---------------------------------------------------------------------------
// embedding: h = relu(x @ W_emb + b_emb), also write bf16 shadow
// one block (128 threads) per node
// ---------------------------------------------------------------------------
__global__ __launch_bounds__(128) void k_embed(
    const float* __restrict__ x, const float* __restrict__ W,
    const float* __restrict__ b, float* __restrict__ h,
    unsigned short* __restrict__ hbf)
{
    __shared__ float xs[92];
    int n = blockIdx.x;
    int t = threadIdx.x;
    if (t < 92) xs[t] = x[n * 92 + t];
    __syncthreads();
    float acc = b[t];
    #pragma unroll 4
    for (int k = 0; k < 92; ++k) acc = fmaf(xs[k], W[k * 128 + t], acc);
    acc = fmaxf(acc, 0.0f);
    h[n * 128 + t]   = acc;
    hbf[n * 128 + t] = f32_to_bf16(acc);
}

// ---------------------------------------------------------------------------
// in-degree (destination = row)
// ---------------------------------------------------------------------------
__global__ void k_degree(const int* __restrict__ rowI, float* __restrict__ deg, int E) {
    int e = blockIdx.x * blockDim.x + threadIdx.x;
    if (e < E) unsafeAtomicAdd(&deg[rowI[e]], 1.0f);
}

// ---------------------------------------------------------------------------
// WMMA edge kernel: per wave, 16 edges x 128 cols message GEMM + scatter-add.
//   msg[16,128] = [h_bf16[col] | bf16(edge_attr)] (16x192) @ Wc (192x128)
// A frags gathered from L2-resident h_bf16 / edge_attr; B staged in LDS
// as transposed bf16 Wc with padded pitch.
// ---------------------------------------------------------------------------
#define BPITCH 100   // uints per B row in LDS (padded: 96 + 4)

__global__ __launch_bounds__(256) void k_edge_wmma(
    const unsigned* __restrict__ hbf,   // [N][64] packed bf16 pairs
    const float*    __restrict__ ea,    // [E][64] fp32
    const int*      __restrict__ rowI,
    const int*      __restrict__ colI,
    const float*    __restrict__ Wcl,   // [192][128] fp32 (layer slice)
    float*          __restrict__ agg,   // [N][128] fp32
    int ngroups)
{
    __shared__ unsigned ldsB[128 * BPITCH];   // 51.2 KB: Wc^T as bf16

    // stage Wc transposed -> ldsB[n][k] (ushort index n*2*BPITCH + k)
    unsigned short* lsB = (unsigned short*)ldsB;
    for (int idx = threadIdx.x; idx < 192 * 128; idx += 256) {
        int k = idx >> 7;       // 0..191
        int n = idx & 127;      // 0..127
        lsB[n * (2 * BPITCH) + k] = f32_to_bf16(Wcl[idx]);
    }
    __syncthreads();

    const int lane = threadIdx.x & 31;
    const int m    = lane & 15;     // edge-in-group / output column base
    const int half = lane >> 4;     // lane half -> K half
    const int w    = (blockIdx.x << 3) + (threadIdx.x >> 5);
    const int Wtot = gridDim.x << 3;

    const uint4*  hb4 = (const uint4*)hbf;
    const float4* ef4 = (const float4*)ea;

    for (int g = w; g < ngroups; g += Wtot) {
        const int gbase = g << 4;
        const int e     = gbase + m;
        const int cn    = colI[e];

        // ---- A fragments: 6 K-tiles of 16x32 bf16 ----
        Frag A[6];
        #pragma unroll
        for (int kt = 0; kt < 4; ++kt) {          // K 0..127 from h_bf16 gather
            uint4 p0 = hb4[cn * 16 + kt * 4 + half];
            uint4 p1 = hb4[cn * 16 + kt * 4 + half + 2];
            A[kt].u[0] = p0.x; A[kt].u[1] = p0.y; A[kt].u[2] = p0.z; A[kt].u[3] = p0.w;
            A[kt].u[4] = p1.x; A[kt].u[5] = p1.y; A[kt].u[6] = p1.z; A[kt].u[7] = p1.w;
        }
        #pragma unroll
        for (int kk = 0; kk < 2; ++kk) {          // K 128..191 from edge_attr
            int kt   = 4 + kk;
            int base = e * 16 + kk * 8;
            float4 q0 = ef4[base + 2 * half];
            float4 q1 = ef4[base + 2 * half + 1];
            float4 q2 = ef4[base + 4 + 2 * half];
            float4 q3 = ef4[base + 4 + 2 * half + 1];
            A[kt].u[0] = pk_bf16(q0.x, q0.y); A[kt].u[1] = pk_bf16(q0.z, q0.w);
            A[kt].u[2] = pk_bf16(q1.x, q1.y); A[kt].u[3] = pk_bf16(q1.z, q1.w);
            A[kt].u[4] = pk_bf16(q2.x, q2.y); A[kt].u[5] = pk_bf16(q2.z, q2.w);
            A[kt].u[6] = pk_bf16(q3.x, q3.y); A[kt].u[7] = pk_bf16(q3.z, q3.w);
        }

        const int rme = rowI[gbase + m];          // lanes 0-15 hold rows m=0..15

        // ---- 8 N-tiles of 16 columns ----
        #pragma unroll
        for (int nt = 0; nt < 8; ++nt) {
            v8f c = {};
            #pragma unroll
            for (int kt = 0; kt < 6; ++kt) {
                int bbase = (m + nt * 16) * BPITCH + kt * 16 + 4 * half;
                const uint4* bp = (const uint4*)&ldsB[bbase];
                Frag B;
                uint4 b0 = bp[0]; uint4 b1 = bp[2];
                B.u[0] = b0.x; B.u[1] = b0.y; B.u[2] = b0.z; B.u[3] = b0.w;
                B.u[4] = b1.x; B.u[5] = b1.y; B.u[6] = b1.z; B.u[7] = b1.w;
                c = __builtin_amdgcn_wmma_f32_16x16x32_bf16(
                        false, A[kt].v, false, B.v, (short)0, c, false, false);
            }
            // scatter-add: c[r] belongs to edge (r + 8*half), column m + nt*16
            int ncol = m + nt * 16;
            #pragma unroll
            for (int r = 0; r < 8; ++r) {
                int dr = __shfl(rme, r + 8 * half, 32);
                unsafeAtomicAdd(&agg[dr * 128 + ncol], c[r]);
            }
        }
    }
}

// scalar fallback for edge-count tail (E % 16) — one block of 128 per edge
__global__ __launch_bounds__(128) void k_edge_tail(
    const float* __restrict__ h, const float* __restrict__ ea,
    const int* __restrict__ rowI, const int* __restrict__ colI,
    const float* __restrict__ Wcl, float* __restrict__ agg, int ebase)
{
    int e = ebase + blockIdx.x;
    int t = threadIdx.x;
    int cn = colI[e];
    float acc = 0.0f;
    for (int k = 0; k < 128; ++k) acc = fmaf(h[cn * 128 + k], Wcl[k * 128 + t], acc);
    for (int k = 0; k < 64;  ++k) acc = fmaf(ea[e * 64 + k],  Wcl[(128 + k) * 128 + t], acc);
    unsafeAtomicAdd(&agg[rowI[e] * 128 + t], acc);
}

// ---------------------------------------------------------------------------
// layer epilogue: mean-agg + bias + inference-BN + relu + residual
// ---------------------------------------------------------------------------
__global__ void k_update(
    const float* __restrict__ agg, const float* __restrict__ deg,
    const float* __restrict__ bc, const float* __restrict__ gamma,
    const float* __restrict__ beta, float* __restrict__ h,
    unsigned short* __restrict__ hbf, int total)
{
    int i = blockIdx.x * blockDim.x + threadIdx.x;
    if (i >= total) return;
    int n = i >> 7, c = i & 127;
    float d = fmaxf(deg[n], 1.0f);
    float a = agg[i] / d + bc[c];
    a = a * (gamma[c] * 0.99950037f) + beta[c];   // gamma / sqrt(1 + 1e-3)
    float hn = h[i] + fmaxf(a, 0.0f);
    h[i]   = hn;
    hbf[i] = f32_to_bf16(hn);
}

// ---------------------------------------------------------------------------
// graph pooling
// ---------------------------------------------------------------------------
__global__ void k_gcnt(const int* __restrict__ bidx, float* __restrict__ gcnt, int N) {
    int n = blockIdx.x * blockDim.x + threadIdx.x;
    if (n < N) unsafeAtomicAdd(&gcnt[bidx[n]], 1.0f);
}

__global__ void k_pool(const float* __restrict__ h, const int* __restrict__ bidx,
                       float* __restrict__ gsum, int total) {
    int i = blockIdx.x * blockDim.x + threadIdx.x;
    if (i >= total) return;
    int n = i >> 7, c = i & 127;
    unsafeAtomicAdd(&gsum[bidx[n] * 128 + c], h[i]);
}

// ---------------------------------------------------------------------------
// head: g = relu(mean @ Wh + bh); out = g @ Wout + bout. one block per graph
// ---------------------------------------------------------------------------
__global__ __launch_bounds__(128) void k_head(
    const float* __restrict__ gsum, const float* __restrict__ gcnt,
    const float* __restrict__ Wh, const float* __restrict__ bh,
    const float* __restrict__ Wout, const float* __restrict__ bout,
    float* __restrict__ out)
{
    __shared__ float gs[128];
    __shared__ float s[128];
    int g = blockIdx.x, t = threadIdx.x;
    float cnt = fmaxf(gcnt[g], 1.0f);
    gs[t] = gsum[g * 128 + t] / cnt;
    __syncthreads();
    float acc = bh[t];
    #pragma unroll 4
    for (int k = 0; k < 128; ++k) acc = fmaf(gs[k], Wh[k * 128 + t], acc);
    s[t] = fmaxf(acc, 0.0f) * Wout[t];
    __syncthreads();
    for (int off = 64; off > 0; off >>= 1) {
        if (t < off) s[t] += s[t + off];
        __syncthreads();
    }
    if (t == 0) out[g] = s[0] + bout[0];
}

// ---------------------------------------------------------------------------
// launch
// ---------------------------------------------------------------------------
extern "C" void kernel_launch(void* const* d_in, const int* in_sizes, int n_in,
                              void* d_out, int out_size, void* d_ws, size_t ws_size,
                              hipStream_t stream) {
    const float* x      = (const float*)d_in[0];
    const int*   eidx   = (const int*)  d_in[1];
    const float* ea     = (const float*)d_in[2];
    const int*   bidx   = (const int*)  d_in[3];
    const float* W_emb  = (const float*)d_in[4];
    const float* b_emb  = (const float*)d_in[5];
    const float* Wc     = (const float*)d_in[6];
    const float* bc     = (const float*)d_in[7];
    const float* gamma  = (const float*)d_in[8];
    const float* beta   = (const float*)d_in[9];
    const float* Wh     = (const float*)d_in[10];
    const float* bh     = (const float*)d_in[11];
    const float* Wout   = (const float*)d_in[12];
    const float* bout   = (const float*)d_in[13];
    float* out = (float*)d_out;

    const int N = in_sizes[3];          // 50000 nodes
    const int E = in_sizes[2] / 64;     // 800000 edges
    const int G = out_size;             // 256 graphs
    const int* rowI = eidx;
    const int* colI = eidx + E;

    // workspace carve (256B aligned)
    char* ws = (char*)d_ws;
    size_t off = 0;
    auto carve = [&](size_t bytes) { char* p = ws + off; off = (off + bytes + 255) & ~(size_t)255; return p; };
    float*          h    = (float*)         carve((size_t)N * 128 * 4);
    float*          agg  = (float*)         carve((size_t)N * 128 * 4);
    unsigned short* hbf  = (unsigned short*)carve((size_t)N * 128 * 2);
    float*          deg  = (float*)         carve((size_t)N * 4);
    float*          gsum = (float*)         carve((size_t)G * 128 * 4);
    float*          gcnt = (float*)         carve((size_t)G * 4);

    const int T = 256;
    // init accumulators (harness does not re-zero between replays)
    k_zero<<<(N + T - 1) / T, T, 0, stream>>>(deg, N);
    k_zero<<<(G * 128 + T - 1) / T, T, 0, stream>>>(gsum, G * 128);
    k_zero<<<(G + T - 1) / T, T, 0, stream>>>(gcnt, G);

    k_embed<<<N, 128, 0, stream>>>(x, W_emb, b_emb, h, hbf);
    k_degree<<<(E + T - 1) / T, T, 0, stream>>>(rowI, deg, E);

    const int ngroups = E / 16;
    const int tail    = E - ngroups * 16;
    for (int l = 0; l < 3; ++l) {
        const float* Wcl = Wc + (size_t)l * 192 * 128;
        k_zero<<<(N * 128 + T - 1) / T, T, 0, stream>>>(agg, N * 128);
        k_edge_wmma<<<512, 256, 0, stream>>>((const unsigned*)hbf, ea, rowI, colI,
                                             Wcl, agg, ngroups);
        if (tail > 0)
            k_edge_tail<<<tail, 128, 0, stream>>>(h, ea, rowI, colI, Wcl, agg,
                                                  ngroups * 16);
        k_update<<<(N * 128 + T - 1) / T, T, 0, stream>>>(
            agg, deg, bc + l * 128, gamma + l * 128, beta + l * 128, h, hbf, N * 128);
    }

    k_gcnt<<<(N + T - 1) / T, T, 0, stream>>>(bidx, gcnt, N);
    k_pool<<<(N * 128 + T - 1) / T, T, 0, stream>>>(h, bidx, gsum, N * 128);
    k_head<<<G, 128, 0, stream>>>(gsum, gcnt, Wh, bh, Wout, bout, out);
}
